// MaxEntIRLNet_26611617366282
// MI455X (gfx1250) — matrix operand order repaired and checked by reading, not compiled, not run on hardware
//
#include <hip/hip_runtime.h>
#include <hip/hip_bf16.h>

typedef __attribute__((ext_vector_type(16))) _Float16 v16h;
typedef __attribute__((ext_vector_type(8)))  _Float16 v8h;
typedef __attribute__((ext_vector_type(8)))  float    v8f;

#define MAXC 384   // widest BN (attention output / score input)

// ---------------------------------------------------------------------------
// small elementwise helpers
// ---------------------------------------------------------------------------
__global__ void cvt_f16_kern(const float* __restrict__ x, _Float16* __restrict__ y, int n) {
    int i = blockIdx.x * blockDim.x + threadIdx.x;
    if (i < n) y[i] = (_Float16)x[i];
}

__global__ void zero_f32_kern(float* __restrict__ p, int n) {
    int i = blockIdx.x * blockDim.x + threadIdx.x;
    if (i < n) p[i] = 0.f;
}

__global__ void add_vec_kern(const float* __restrict__ a, const float* __restrict__ b,
                             float* __restrict__ o, int n) {
    int i = blockIdx.x * blockDim.x + threadIdx.x;
    if (i < n) o[i] = a[i] + b[i];
}

// ---------------------------------------------------------------------------
// BatchNorm batch statistics: two-stage reduction.
// Stage 1: row-tiled, fully coalesced global reads; per-column partials built
// with LDS float atomics (ds_add_f32), flushed with global f32 atomics.
// ---------------------------------------------------------------------------
template <typename T>
__global__ void stats_partial_kern(const T* __restrict__ x, float* __restrict__ acc,
                                   int C, int rowsPerBlk) {
    __shared__ float ls[2 * MAXC];
    const int tid = threadIdx.x;
    for (int i = tid; i < 2 * C; i += blockDim.x) ls[i] = 0.f;
    __syncthreads();
    const size_t base = (size_t)blockIdx.x * rowsPerBlk * C;
    const int total = rowsPerBlk * C;
    for (int i = tid; i < total; i += blockDim.x) {
        float v = (float)x[base + i];
        int c = i % C;
        atomicAdd(&ls[c], v);
        atomicAdd(&ls[C + c], v * v);
    }
    __syncthreads();
    for (int i = tid; i < 2 * C; i += blockDim.x) atomicAdd(&acc[i], ls[i]);
}

// Stage 2: fold gamma/beta into an affine (s, t):  y = x*s + t
__global__ void stats_finalize_kern(const float* __restrict__ acc,
                                    const float* __restrict__ g, const float* __restrict__ b,
                                    float* __restrict__ s, float* __restrict__ t,
                                    int C, float invB) {
    int c = blockIdx.x * blockDim.x + threadIdx.x;
    if (c < C) {
        float m   = acc[c] * invB;
        float var = acc[C + c] * invB - m * m;
        float sc  = g[c] * rsqrtf(var + 1e-5f);
        s[c] = sc;
        t[c] = b[c] - m * sc;
    }
}

template <typename T>
__global__ void bn_apply_kern(const T* __restrict__ x, const float* __restrict__ s,
                              const float* __restrict__ t, _Float16* __restrict__ y,
                              int C, size_t total) {
    size_t i = (size_t)blockIdx.x * blockDim.x + threadIdx.x;
    if (i < total) {
        int c = (int)(i % (size_t)C);
        y[i] = (_Float16)((float)x[i] * s[c] + t[c]);
    }
}

// ---------------------------------------------------------------------------
// WMMA GEMM:  C[M,N] = act( A[M,K] @ W[N,K]^T + bias[N] ),  f16 in/out, f32
// accumulate via v_wmma_f32_16x16x32_f16.
// Tiling: block = 256 threads = 8 waves; each wave owns a 32-row (MT=2) x
// NT*16-col tile.  Each B fragment feeds TWO WMMAs (A-tile pair), so per
// k-step: 12 b128 loads -> 8 WMMAs, with only 4 B fragments live (32 VGPRs).
// K is a template constant and the k-loop fully unrolls: every load address
// is base + immediate offset (24-bit IOFFSET), so the hot code has no address
// VALU between WMMAs (no hazard NOPs) and the scheduler can hoist loads and
// emit descending s_wait_loadcnt.
// Fragment layouts per CDNA5 ISA 7.12.2 (16-bit A 16x32; B column-per-lane;
// C/D 8 VGPRs: row = hi*8 + r, col = lo).
// ---------------------------------------------------------------------------
template <int K, int NT, bool RELU>
__global__ void gemm_wmma_kern(const _Float16* __restrict__ A, const _Float16* __restrict__ W,
                               const float* __restrict__ bias, _Float16* __restrict__ C,
                               int N) {
    const int lane   = threadIdx.x & 31;
    const int wave   = threadIdx.x >> 5;
    const int lo     = lane & 15;
    const int hi     = lane >> 4;
    const int mBase  = (blockIdx.x * 8 + wave) * 32;           // MT=2 row tiles
    const int nChunk = blockIdx.y * (NT * 16);

    v8f acc[2][NT];
    #pragma unroll
    for (int m = 0; m < 2; ++m)
        #pragma unroll
        for (int j = 0; j < NT; ++j) acc[m][j] = (v8f){};

    const _Float16* aptr0 = A + (size_t)(mBase + lo) * K + hi * 16;
    const _Float16* aptr1 = aptr0 + (size_t)16 * K;
    const _Float16* wptr  = W + (size_t)(nChunk + lo) * K + hi * 16;

    #pragma unroll
    for (int k0 = 0; k0 < K; k0 += 32) {
        // ---- load phase: 2 A fragments + NT shared B fragments ----
        union { v16h v; v8h h[2]; } a0, a1;
        a0.h[0] = *(const v8h*)(aptr0 + k0);
        a0.h[1] = *(const v8h*)(aptr0 + k0 + 8);
        a1.h[0] = *(const v8h*)(aptr1 + k0);
        a1.h[1] = *(const v8h*)(aptr1 + k0 + 8);
        v16h bf[NT];
        #pragma unroll
        for (int j = 0; j < NT; ++j)
            bf[j] = *(const v16h*)(wptr + j * 16 * K + k0);

        // ---- WMMA phase: each B fragment reused by both A tiles ----
        #pragma unroll
        for (int j = 0; j < NT; ++j) {
            acc[0][j] = __builtin_amdgcn_wmma_f32_16x16x32_f16(
                false, a0.v, false, bf[j], (short)0, acc[0][j], false, false);
            acc[1][j] = __builtin_amdgcn_wmma_f32_16x16x32_f16(
                false, a1.v, false, bf[j], (short)0, acc[1][j], false, false);
        }
    }

    #pragma unroll
    for (int m = 0; m < 2; ++m) {
        #pragma unroll
        for (int j = 0; j < NT; ++j) {
            int   n  = nChunk + j * 16 + lo;
            float bv = bias[n];
            #pragma unroll
            for (int r = 0; r < 8; ++r) {
                int row = mBase + m * 16 + hi * 8 + r;
                float v = acc[m][j][r] + bv;
                if (RELU) v = v > 0.f ? v : 0.f;
                C[(size_t)row * N + n] = (_Float16)v;
            }
        }
    }
}

// ---------------------------------------------------------------------------
// LSTM cell (h0=c0=0): gates [i|f|c|o] of width 64; f is dead.
// h = sigmoid(o) * tanh( sigmoid(i) * tanh(c) ).  Writes into concat buffer.
// ---------------------------------------------------------------------------
__global__ void lstm_act_kern(const _Float16* __restrict__ g, _Float16* __restrict__ xcat,
                              int colOff, int B) {
    int i = blockIdx.x * blockDim.x + threadIdx.x;
    if (i < B * 64) {
        int row = i >> 6, j = i & 63;
        const _Float16* gr = g + (size_t)row * 256;
        float gi = (float)gr[j];
        float gc = (float)gr[128 + j];
        float go = (float)gr[192 + j];
        float cell = (1.f / (1.f + __expf(-gi))) * tanhf(gc);
        float h    = (1.f / (1.f + __expf(-go))) * tanhf(cell);
        xcat[(size_t)row * 384 + colOff + j] = (_Float16)h;
    }
}

// Final 32->1 layer: plain dot product (WMMA would waste 15/16 of a tile).
__global__ void final_dot_kern(const _Float16* __restrict__ a, const float* __restrict__ w,
                               const float* __restrict__ b3, float* __restrict__ out, int B) {
    int row = blockIdx.x * blockDim.x + threadIdx.x;
    if (row < B) {
        const _Float16* ar = a + (size_t)row * 32;
        float s = 0.f;
        #pragma unroll
        for (int k = 0; k < 32; ++k) s += (float)ar[k] * w[k];
        out[row] = s + b3[0];
    }
}

// ---------------------------------------------------------------------------
// Orchestration
// ---------------------------------------------------------------------------
extern "C" void kernel_launch(void* const* d_in, const int* in_sizes, int n_in,
                              void* d_out, int out_size, void* d_ws, size_t ws_size,
                              hipStream_t stream) {
    (void)n_in; (void)out_size; (void)ws_size;
    const int dims[6] = {64, 64, 96, 160, 32, 32};
    const int B = in_sizes[0] / 64;              // 65536
    const float invB = 1.0f / (float)B;

    // ---- input index map (setup_inputs() dict order, recursed in-order) ----
    // 0..5 : ttc, acc_info, dynamics, surroundings, lane_change, overtaking
    // 6+9n : bn1_g, bn1_b, W, b, bn2_g, bn2_b, Wih, bih, bhh   (branch n)
    // 60   : Wv, bv, Wo, bo
    // 64   : bn0_g, bn0_b, W1, b1, bn1_g, bn1_b, W2, b2, bn2_g, bn2_b, W3, b3
    auto F = [&](int i) { return (const float*)d_in[i]; };

    // ---- workspace arena ----
    char* wsb = (char*)d_ws;
    size_t off = 0;
    auto sub = [&](size_t bytes) -> char* {
        char* p = wsb + off;
        off += (bytes + 255) & ~(size_t)255;
        return p;
    };
    _Float16* Abuf = (_Float16*)sub((size_t)B * 384 * 2);
    _Float16* Bbuf = (_Float16*)sub((size_t)B * 384 * 2);
    _Float16* Xcat = (_Float16*)sub((size_t)B * 384 * 2);
    _Float16 *Wf[6], *Wih[6];
    for (int n = 0; n < 6; ++n) {
        Wf[n]  = (_Float16*)sub((size_t)dims[n] * dims[n] * 2);
        Wih[n] = (_Float16*)sub((size_t)256 * dims[n] * 2);
    }
    _Float16* Wv16 = (_Float16*)sub((size_t)384 * 384 * 2);
    _Float16* Wo16 = (_Float16*)sub((size_t)384 * 384 * 2);
    _Float16* W116 = (_Float16*)sub((size_t)64 * 384 * 2);
    _Float16* W216 = (_Float16*)sub((size_t)32 * 64 * 2);
    float* acc  = (float*)sub(2 * MAXC * 4);
    float* sArr = (float*)sub(MAXC * 4);
    float* tArr = (float*)sub(MAXC * 4);
    float* bsum = (float*)sub(256 * 4);

    auto cvt = [&](const float* src, _Float16* dst, int n) {
        cvt_f16_kern<<<(n + 255) / 256, 256, 0, stream>>>(src, dst, n);
    };
    // NT chosen so N % (NT*16) == 0 -> branch-free k-loop.  MT=2 rows/wave.
    auto gemm = [&](bool relu, const _Float16* A, const _Float16* W, const float* bias,
                    _Float16* C, int K, int N) {
        int nt = (N % 64 == 0) ? 4 : (N % 48 == 0) ? 3 : 2;
        dim3 grid(B / 256, N / (nt * 16));
        if (relu) {
            if      (K == 384)            gemm_wmma_kern<384, 4, true ><<<grid, 256, 0, stream>>>(A, W, bias, C, N);
            else if (K == 160)            gemm_wmma_kern<160, 2, true ><<<grid, 256, 0, stream>>>(A, W, bias, C, N);
            else if (K == 96)             gemm_wmma_kern< 96, 3, true ><<<grid, 256, 0, stream>>>(A, W, bias, C, N);
            else if (K == 64 && N == 64)  gemm_wmma_kern< 64, 4, true ><<<grid, 256, 0, stream>>>(A, W, bias, C, N);
            else if (K == 64)             gemm_wmma_kern< 64, 2, true ><<<grid, 256, 0, stream>>>(A, W, bias, C, N);
            else                          gemm_wmma_kern< 32, 2, true ><<<grid, 256, 0, stream>>>(A, W, bias, C, N);
        } else {   // N == 256 (LSTM) or N == 384 (attention): always NT=4
            if      (K == 384)            gemm_wmma_kern<384, 4, false><<<grid, 256, 0, stream>>>(A, W, bias, C, N);
            else if (K == 160)            gemm_wmma_kern<160, 4, false><<<grid, 256, 0, stream>>>(A, W, bias, C, N);
            else if (K == 96)             gemm_wmma_kern< 96, 4, false><<<grid, 256, 0, stream>>>(A, W, bias, C, N);
            else if (K == 64)             gemm_wmma_kern< 64, 4, false><<<grid, 256, 0, stream>>>(A, W, bias, C, N);
            else                          gemm_wmma_kern< 32, 4, false><<<grid, 256, 0, stream>>>(A, W, bias, C, N);
        }
    };
    const int NB = 512;            // stats stage-1 blocks; B/NB rows each
    auto stats_f32 = [&](const float* x, int C, const float* g, const float* b) {
        zero_f32_kern<<<(2 * C + 255) / 256, 256, 0, stream>>>(acc, 2 * C);
        stats_partial_kern<float><<<NB, 256, 0, stream>>>(x, acc, C, B / NB);
        stats_finalize_kern<<<(C + 255) / 256, 256, 0, stream>>>(acc, g, b, sArr, tArr, C, invB);
    };
    auto stats_f16 = [&](const _Float16* x, int C, const float* g, const float* b) {
        zero_f32_kern<<<(2 * C + 255) / 256, 256, 0, stream>>>(acc, 2 * C);
        stats_partial_kern<_Float16><<<NB, 256, 0, stream>>>(x, acc, C, B / NB);
        stats_finalize_kern<<<(C + 255) / 256, 256, 0, stream>>>(acc, g, b, sArr, tArr, C, invB);
    };
    auto bn_f32 = [&](const float* x, _Float16* y, int C) {
        size_t tot = (size_t)B * C;
        bn_apply_kern<float><<<(unsigned)((tot + 255) / 256), 256, 0, stream>>>(x, sArr, tArr, y, C, tot);
    };
    auto bn_f16 = [&](const _Float16* x, _Float16* y, int C) {
        size_t tot = (size_t)B * C;
        bn_apply_kern<_Float16><<<(unsigned)((tot + 255) / 256), 256, 0, stream>>>(x, sArr, tArr, y, C, tot);
    };

    // ---- one-time weight conversion (f32 -> f16) ----
    for (int n = 0; n < 6; ++n) {
        cvt(F(6 + 9 * n + 2), Wf[n],  dims[n] * dims[n]);
        cvt(F(6 + 9 * n + 6), Wih[n], 256 * dims[n]);
    }
    cvt(F(60), Wv16, 384 * 384);
    cvt(F(62), Wo16, 384 * 384);
    cvt(F(64 + 2), W116, 64 * 384);
    cvt(F(64 + 6), W216, 32 * 64);

    // ---- six branches: BN -> Linear+ReLU -> BN -> LSTM-cell -> concat ----
    for (int n = 0; n < 6; ++n) {
        const int d = dims[n];
        const int p = 6 + 9 * n;
        stats_f32(F(n), d, F(p + 0), F(p + 1));
        bn_f32(F(n), Abuf, d);
        gemm(true, Abuf, Wf[n], F(p + 3), Bbuf, d, d);
        stats_f16(Bbuf, d, F(p + 4), F(p + 5));
        bn_f16(Bbuf, Abuf, d);
        add_vec_kern<<<1, 256, 0, stream>>>(F(p + 7), F(p + 8), bsum, 256);  // bih + bhh
        gemm(false, Abuf, Wih[n], bsum, Bbuf, d, 256);
        lstm_act_kern<<<(B * 64 + 255) / 256, 256, 0, stream>>>(Bbuf, Xcat, 64 * n, B);
    }

    // ---- degenerate attention (seq_len==1): out_proj(v_proj(x)) ----
    gemm(false, Xcat, Wv16, F(61), Abuf, 384, 384);
    gemm(false, Abuf, Wo16, F(63), Bbuf, 384, 384);

    // ---- score head ----
    const int s0 = 64;
    stats_f16(Bbuf, 384, F(s0 + 0), F(s0 + 1));
    bn_f16(Bbuf, Abuf, 384);
    gemm(true, Abuf, W116, F(s0 + 3), Bbuf, 384, 64);
    stats_f16(Bbuf, 64, F(s0 + 4), F(s0 + 5));
    bn_f16(Bbuf, Abuf, 64);
    gemm(true, Abuf, W216, F(s0 + 7), Bbuf, 64, 32);
    stats_f16(Bbuf, 32, F(s0 + 8), F(s0 + 9));
    bn_f16(Bbuf, Abuf, 32);
    final_dot_kern<<<(B + 255) / 256, 256, 0, stream>>>(Abuf, F(s0 + 10), F(s0 + 11),
                                                        (float*)d_out, B);
}
